// ECGModel_18081812316628
// MI455X (gfx1250) — matrix operand order repaired and checked by reading, not compiled, not run on hardware
//
#include <hip/hip_runtime.h>

// ---------------------------------------------------------------------------
// MI455X (gfx1250) implementation. All large matmuls use
// v_wmma_f32_16x16x32_f16 (f16 in, f32 accumulate); weights pre-converted to
// f16 once per launch so GEMM inner loops are cvt-free. Attention is
// flash-style with an LDS-transposed softmax (no cross-lane bpermutes).
// wave32 operand layouts per CDNA5 ISA 7.12.2.
// ---------------------------------------------------------------------------

typedef __attribute__((ext_vector_type(16))) _Float16 v16h;
typedef __attribute__((ext_vector_type(8)))  float    v8f;

constexpr int kB = 8, kN = 1024, kF = 16, kD = 128, kDFF = 2048;
constexpr int kH = 4, kL = 2, kG = 2, kK = 6, kDH = 32;
constexpr int kM = kB * kN;          // 8192 token rows

__device__ __forceinline__ float gelu_f(float x) {
  return 0.5f * x * (1.0f + erff(x * 0.70710678118654752f));
}

__global__ void f32_to_f16_kernel(const float* __restrict__ src,
                                  _Float16* __restrict__ dst, int n) {
  int i = blockIdx.x * 256 + threadIdx.x;
  if (i < n) dst[i] = (_Float16)src[i];
}

// ---------------------------------------------------------------------------
// Generic GEMM: C[M x O] = act(A[M x I] @ W16[O x I]^T + bias) (+C if ACC)
// Block = 128 threads (4 waves); block tile 64(M) x 64(O); each wave owns a
// 16-row strip and 4 16x16 C tiles. K-loop steps of 32 (one WMMA each tile).
// W is pre-converted f16 -> B operand loads are two b128 per tile, no cvt.
// ACT: 0=none, 1=relu, 2=gelu.
// ---------------------------------------------------------------------------
template <typename AT, typename OT, int ACT, bool ACC>
__launch_bounds__(128)
__global__ void gemm_wmma_kernel(const AT* __restrict__ A,
                                 const _Float16* __restrict__ W,
                                 const float* __restrict__ bias,
                                 OT* __restrict__ C,
                                 int Mtot, int I, int O) {
  const int lane = threadIdx.x & 31;
  const int wave = threadIdx.x >> 5;
  const int n16  = lane & 15;
  const int hi   = lane >> 4;
  const int m0   = blockIdx.y * 64 + wave * 16;
  const int o0   = blockIdx.x * 64;

  v8f acc[4];
#pragma unroll
  for (int t = 0; t < 4; t++)
#pragma unroll
    for (int j = 0; j < 8; j++)
      acc[t][j] = ACC ? ((const float*)(const void*)C)
                            [(size_t)(m0 + j + 8 * hi) * O + (o0 + t * 16 + n16)]
                      : 0.0f;

  const AT* arow = A + (size_t)(m0 + n16) * I;
  for (int k0 = 0; k0 < I; k0 += 32) {
    // A operand: lane = row; elems 0..15 -> K = k0+hi*16.. (contiguous)
    v16h a;
#pragma unroll
    for (int i = 0; i < 8; i++) a[i]     = (_Float16)(float)arow[k0 + hi * 16 + i];
#pragma unroll
    for (int i = 0; i < 8; i++) a[8 + i] = (_Float16)(float)arow[k0 + hi * 16 + 8 + i];
    if (k0 + 32 < I) __builtin_prefetch(arow + k0 + 32, 0, 0);  // global_prefetch_b8
#pragma unroll
    for (int t = 0; t < 4; t++) {
      // B operand: lane = output column; contiguous 16 k-values of W row (f16).
      const _Float16* wr = W + (size_t)(o0 + t * 16 + n16) * I + k0 + hi * 16;
      v16h bm;
#pragma unroll
      for (int i = 0; i < 16; i++) bm[i] = wr[i];
      if (k0 + 32 < I) __builtin_prefetch(wr + 32, 0, 0);
      acc[t] = __builtin_amdgcn_wmma_f32_16x16x32_f16(false, a, false, bm,
                                                      (short)0, acc[t], false, false);
    }
  }
#pragma unroll
  for (int t = 0; t < 4; t++) {
    const int oc = o0 + t * 16 + n16;
    const float bv = bias[oc];
#pragma unroll
    for (int j = 0; j < 8; j++) {
      float v = acc[t][j] + bv;
      if (ACT == 1) v = fmaxf(v, 0.0f);
      if (ACT == 2) v = gelu_f(v);
      C[(size_t)(m0 + j + 8 * hi) * O + oc] = (OT)v;
    }
  }
}

// ---------------------------------------------------------------------------
// Flash attention over f16 QKV. qkv layout: [(b*N+n)*3D] halfs; q at +h*DH,
// k at +D+h*DH, v at +2D+h*DH. One wave = 16 query rows; block = 4 waves.
// Softmax: masked scores go through LDS (C-layout -> row-major), every lane
// re-reads its full 32-wide row and reduces in registers -- no bpermutes.
// Per-row rescale factors return to C-layout lanes via a 16-entry LDS array.
// ---------------------------------------------------------------------------
__launch_bounds__(128)
__global__ void attn_kernel(const _Float16* __restrict__ qkv,
                            const unsigned char* __restrict__ valid,
                            float* __restrict__ out) {
  __shared__ float sbuf[4][16][32];
  __shared__ float scl[4][16];
  __shared__ float lbuf[4][16];
  const int lane = threadIdx.x & 31;
  const int wave = threadIdx.x >> 5;
  const int n16 = lane & 15, hi = lane >> 4;
  const int b = blockIdx.z, h = blockIdx.y;
  const int q0 = blockIdx.x * 64 + wave * 16;
  const size_t bh = (size_t)b * kN;
  const float qscale = 0.17677669529663687f;  // 1/sqrt(DH)

  const _Float16* qp = qkv + (bh + (size_t)(q0 + n16)) * (3 * kD) + h * kDH;
  v16h aq;
#pragma unroll
  for (int i = 0; i < 8; i++) { aq[i] = qp[hi * 16 + i]; aq[8 + i] = qp[hi * 16 + 8 + i]; }

  float mrow = -3.0e38f, lrow = 0.f;   // stats for row n16 (dup in both halves)
  v8f acc0, acc1;
#pragma unroll
  for (int j = 0; j < 8; j++) { acc0[j] = 0.f; acc1[j] = 0.f; }

  for (int kc = 0; kc < kN; kc += 32) {
    const int col0 = kc + n16, col1 = kc + 16 + n16;
    const _Float16* kp0 = qkv + (bh + col0) * (3 * kD) + kD + h * kDH + hi * 16;
    const _Float16* kp1 = qkv + (bh + col1) * (3 * kD) + kD + h * kDH + hi * 16;
    v16h bk0, bk1;
#pragma unroll
    for (int i = 0; i < 16; i++) { bk0[i] = kp0[i]; bk1[i] = kp1[i]; }
    v8f s0, s1;
#pragma unroll
    for (int j = 0; j < 8; j++) { s0[j] = 0.f; s1[j] = 0.f; }
    s0 = __builtin_amdgcn_wmma_f32_16x16x32_f16(false, aq, false, bk0, (short)0, s0, false, false);
    s1 = __builtin_amdgcn_wmma_f32_16x16x32_f16(false, aq, false, bk1, (short)0, s1, false, false);
    const bool v0 = valid[bh + col0] != 0;
    const bool v1 = valid[bh + col1] != 0;
#pragma unroll
    for (int j = 0; j < 8; j++) {
      sbuf[wave][j + 8 * hi][n16]      = v0 ? s0[j] * qscale : -1.0e9f;
      sbuf[wave][j + 8 * hi][16 + n16] = v1 ? s1[j] * qscale : -1.0e9f;
    }
    asm volatile("s_wait_dscnt 0" ::: "memory");  // wave-local LDS RAW
    float srow[32];
#pragma unroll
    for (int c = 0; c < 32; c++) srow[c] = sbuf[wave][n16][c];
    float rmax = srow[0];
#pragma unroll
    for (int c = 1; c < 32; c++) rmax = fmaxf(rmax, srow[c]);
    const float mnew = fmaxf(mrow, rmax);
    const float sc = __expf(mrow - mnew);
    float rsum = 0.f;
#pragma unroll
    for (int c = 0; c < 32; c++) { srow[c] = __expf(srow[c] - mnew); rsum += srow[c]; }
    mrow = mnew;
    lrow = lrow * sc + rsum;
    if (lane < 16) scl[wave][n16] = sc;
    asm volatile("s_wait_dscnt 0" ::: "memory");
    // P already row-major per lane -> A operand directly from registers
    v16h ap;
#pragma unroll
    for (int i = 0; i < 8; i++) {
      ap[i]     = (_Float16)srow[hi * 16 + i];
      ap[8 + i] = (_Float16)srow[hi * 16 + 8 + i];
    }
#pragma unroll
    for (int j = 0; j < 8; j++) {
      const float s = scl[wave][j + 8 * hi];
      acc0[j] *= s; acc1[j] *= s;
    }
    v16h bv0, bv1;
#pragma unroll
    for (int i = 0; i < 16; i++) {
      const _Float16* vp = qkv + (bh + (size_t)(kc + hi * 16 + i)) * (3 * kD) + 2 * kD + h * kDH;
      bv0[i] = vp[n16];
      bv1[i] = vp[16 + n16];
    }
    acc0 = __builtin_amdgcn_wmma_f32_16x16x32_f16(false, ap, false, bv0, (short)0, acc0, false, false);
    acc1 = __builtin_amdgcn_wmma_f32_16x16x32_f16(false, ap, false, bv1, (short)0, acc1, false, false);
  }
  if (lane < 16) lbuf[wave][n16] = lrow;
  asm volatile("s_wait_dscnt 0" ::: "memory");
#pragma unroll
  for (int j = 0; j < 8; j++) {
    const float inv = 1.0f / lbuf[wave][j + 8 * hi];
    float* op = out + (bh + (size_t)(q0 + j + 8 * hi)) * kD + h * kDH;
    op[n16]      = acc0[j] * inv;
    op[16 + n16] = acc1[j] * inv;
  }
}

// ---------------------------------------------------------------------------
// Small scalar kernels
// ---------------------------------------------------------------------------
__global__ void encoder_kernel(const float* __restrict__ beats,
                               const float* __restrict__ w1, const float* __restrict__ b1,
                               const float* __restrict__ w2, const float* __restrict__ b2,
                               float* __restrict__ e) {
  const int row = blockIdx.x, t = threadIdx.x;
  __shared__ float hsh[64];
  if (t < 64) {
    float s = b1[t];
    for (int f = 0; f < kF; f++) s += beats[(size_t)row * kF + f] * w1[t * kF + f];
    hsh[t] = gelu_f(s);
  }
  __syncthreads();
  float s = b2[t];
  for (int j = 0; j < 64; j++) s += hsh[j] * w2[t * 64 + j];
  e[(size_t)row * kD + t] = s;
}

__global__ void mask_kernel(const float* __restrict__ e,
                            const unsigned char* __restrict__ nm,
                            const unsigned char* __restrict__ vm,
                            const float* __restrict__ mtok, float* __restrict__ o) {
  const int row = blockIdx.x, d = threadIdx.x;
  const bool rep = nm[row] && vm[row];
  float v = rep ? mtok[d] : e[(size_t)row * kD + d];
  o[(size_t)row * kD + d] = vm[row] ? v : 0.0f;
}

// y = LN((do_gelu ? gelu(a) : a) + r) * g + bt, over D=128 (one row per block)
__global__ void resln_kernel(const float* __restrict__ a, const float* __restrict__ r,
                             const float* __restrict__ g, const float* __restrict__ bt,
                             float* __restrict__ y, int do_gelu) {
  const int row = blockIdx.x, d = threadIdx.x;
  __shared__ float red[128];
  float av = a[(size_t)row * kD + d];
  float v = (do_gelu ? gelu_f(av) : av) + r[(size_t)row * kD + d];
  red[d] = v; __syncthreads();
  for (int s = 64; s > 0; s >>= 1) { if (d < s) red[d] += red[d + s]; __syncthreads(); }
  const float mean = red[0] / (float)kD; __syncthreads();
  red[d] = (v - mean) * (v - mean); __syncthreads();
  for (int s = 64; s > 0; s >>= 1) { if (d < s) red[d] += red[d + s]; __syncthreads(); }
  const float var = red[0] / (float)kD;
  y[(size_t)row * kD + d] = (v - mean) * rsqrtf(var + 1e-5f) * g[d] + bt[d];
}

__global__ void xnorm_kernel(const float* __restrict__ x, float* __restrict__ xn) {
  const int row = blockIdx.x, d = threadIdx.x;
  __shared__ float red[128];
  float v = x[(size_t)row * kD + d];
  red[d] = v * v; __syncthreads();
  for (int s = 64; s > 0; s >>= 1) { if (d < s) red[d] += red[d + s]; __syncthreads(); }
  float nrm = fmaxf(sqrtf(red[0]), 1e-12f);
  xn[(size_t)row * kD + d] = v / nrm;
}

// top-(K+1) cosine neighbors per node + chain + self -> neighbor index list
__global__ void topk_kernel(const float* __restrict__ xn, int* __restrict__ nbr) {
  const int n = blockIdx.x, b = blockIdx.y, t = threadIdx.x;
  __shared__ float qv[128];
  __shared__ float sval[128];
  __shared__ int sidx[128];
  const float* base = xn + (size_t)b * kN * kD;
  qv[t] = base[(size_t)n * kD + t];
  __syncthreads();
  float simv[8];
  int selected = 0;
  for (int i = 0; i < 8; i++) {
    const float* xr = base + (size_t)(t + i * 128) * kD;
    float s = 0.f;
    for (int d = 0; d < kD; d++) s += qv[d] * xr[d];
    simv[i] = s;
  }
  int picks[kK + 1];
  for (int p = 0; p < kK + 1; p++) {
    float best = -3.0e38f; int bidx = kN;
    for (int i = 0; i < 8; i++) {
      if (!((selected >> i) & 1)) {
        int m = t + i * 128;
        if (simv[i] > best) { best = simv[i]; bidx = m; }
      }
    }
    sval[t] = best; sidx[t] = bidx; __syncthreads();
    for (int s = 64; s > 0; s >>= 1) {
      if (t < s) {
        float ov = sval[t + s]; int oi = sidx[t + s];
        if (ov > sval[t] || (ov == sval[t] && oi < sidx[t])) { sval[t] = ov; sidx[t] = oi; }
      }
      __syncthreads();
    }
    const int win = sidx[0];
    picks[p] = win;
    if ((win & 127) == t) selected |= 1 << (win >> 7);
    __syncthreads();
  }
  if (t == 0) {
    int list[12]; int cnt = 0;
    list[cnt++] = n;                                   // eye
    for (int p = 0; p < kK + 1; p++) if (picks[p] != n) list[cnt++] = picks[p];
    for (int dd = -1; dd <= 1; dd += 2) {              // chain
      int m = n + dd;
      if (m < 0 || m >= kN) continue;
      bool found = false;
      for (int i = 0; i < cnt; i++) if (list[i] == m) found = true;
      if (!found) list[cnt++] = m;
    }
    int* o = nbr + (size_t)(b * kN + n) * 12;
    o[0] = cnt;
    for (int i = 0; i < cnt; i++) o[1 + i] = list[i];
    for (int i = cnt; i < 11; i++) o[1 + i] = 0;
  }
}

__global__ void agg_kernel(const float* __restrict__ x, const int* __restrict__ nbr,
                           float* __restrict__ agg) {
  const int row = blockIdx.x, d = threadIdx.x;
  const int b = row / kN;
  const int* nb = nbr + (size_t)row * 12;
  const int cnt = nb[0];
  float s = 0.f;
  for (int i = 0; i < cnt; i++)
    s += x[((size_t)b * kN + nb[1 + i]) * kD + d];
  agg[(size_t)row * kD + d] = s / ((float)cnt + 1e-6f);
}

// ---------------------------------------------------------------------------
// Host orchestration
// ---------------------------------------------------------------------------
struct TW {
  const _Float16 *qkv_w, *out_w, *ff1_w, *ff2_w;              // f16 copies
  const float *qkv_b, *out_b, *ff1_b, *ff2_b;
  const float *ln1_g, *ln1_b, *ln2_g, *ln2_b;
};

static void run_temporal(float* x, float* xalt, _Float16* qkvh, float* attno, float* tmp,
                         _Float16* hff, const unsigned char* valid, const TW& w,
                         hipStream_t stream) {
  for (int l = 0; l < kL; l++) {
    gemm_wmma_kernel<float, _Float16, 0, false><<<dim3(3 * kD / 64, kM / 64), 128, 0, stream>>>(
        x, w.qkv_w + (size_t)l * 3 * kD * kD, w.qkv_b + l * 3 * kD, qkvh, kM, kD, 3 * kD);
    attn_kernel<<<dim3(kN / 64, kH, kB), 128, 0, stream>>>(qkvh, valid, attno);
    gemm_wmma_kernel<float, float, 0, false><<<dim3(kD / 64, kM / 64), 128, 0, stream>>>(
        attno, w.out_w + (size_t)l * kD * kD, w.out_b + l * kD, tmp, kM, kD, kD);
    resln_kernel<<<kM, 128, 0, stream>>>(tmp, x, w.ln1_g + l * kD, w.ln1_b + l * kD, xalt, 0);
    gemm_wmma_kernel<float, _Float16, 1, false><<<dim3(kDFF / 64, kM / 64), 128, 0, stream>>>(
        xalt, w.ff1_w + (size_t)l * kDFF * kD, w.ff1_b + l * kDFF, hff, kM, kD, kDFF);
    gemm_wmma_kernel<_Float16, float, 0, false><<<dim3(kD / 64, kM / 64), 128, 0, stream>>>(
        hff, w.ff2_w + (size_t)l * kD * kDFF, w.ff2_b + l * kD, tmp, kM, kDFF, kD);
    resln_kernel<<<kM, 128, 0, stream>>>(tmp, xalt, w.ln2_g + l * kD, w.ln2_b + l * kD, x, 0);
  }
}

extern "C" void kernel_launch(void* const* d_in, const int* in_sizes, int n_in,
                              void* d_out, int out_size, void* d_ws, size_t ws_size,
                              hipStream_t stream) {
  const float* beats = (const float*)d_in[0];
  /* d_in[1] = rr (unused by reference) */
  const unsigned char* node_mask = (const unsigned char*)d_in[2];  // numpy bool = 1 byte
  const unsigned char* valid     = (const unsigned char*)d_in[3];
  const float* enc_w1 = (const float*)d_in[4];
  const float* enc_b1 = (const float*)d_in[5];
  const float* enc_w2 = (const float*)d_in[6];
  const float* enc_b2 = (const float*)d_in[7];
  const float* t_qkv_w = (const float*)d_in[8];
  const float* t_qkv_b = (const float*)d_in[9];
  const float* t_out_w = (const float*)d_in[10];
  const float* t_out_b = (const float*)d_in[11];
  const float* t_ff1_w = (const float*)d_in[12];
  const float* t_ff1_b = (const float*)d_in[13];
  const float* t_ff2_w = (const float*)d_in[14];
  const float* t_ff2_b = (const float*)d_in[15];
  const float* t_ln1_g = (const float*)d_in[16];
  const float* t_ln1_b = (const float*)d_in[17];
  const float* t_ln2_g = (const float*)d_in[18];
  const float* t_ln2_b = (const float*)d_in[19];
  const float* g_self_w  = (const float*)d_in[20];
  const float* g_self_b  = (const float*)d_in[21];
  const float* g_neigh_w = (const float*)d_in[22];
  const float* g_neigh_b = (const float*)d_in[23];
  const float* g_ln_g    = (const float*)d_in[24];
  const float* g_ln_b    = (const float*)d_in[25];
  const float* dec_w     = (const float*)d_in[26];
  const float* dec_b     = (const float*)d_in[27];
  const float* mask_tok  = (const float*)d_in[28];
  float* out = (float*)d_out;  // [target (M*D) | recon (M*D)]

  // workspace partition (~70 MB)
  char* ws = (char*)d_ws;
  size_t off = 0;
  auto carve = [&](size_t bytes) -> void* {
    void* p = ws + off;
    off += (bytes + 255) & ~(size_t)255;
    return p;
  };
  float*     e     = (float*)carve((size_t)kM * kD * 4);
  float*     xa    = (float*)carve((size_t)kM * kD * 4);
  float*     xb    = (float*)carve((size_t)kM * kD * 4);
  _Float16*  qkvh  = (_Float16*)carve((size_t)kM * 3 * kD * 2);
  float*     attno = (float*)carve((size_t)kM * kD * 4);
  float*     tmp   = (float*)carve((size_t)kM * kD * 4);
  _Float16*  hff   = (_Float16*)carve((size_t)kM * kDFF * 2);
  float*     xn    = (float*)carve((size_t)kM * kD * 4);
  float*     aggb  = (float*)carve((size_t)kM * kD * 4);
  float*     gout  = (float*)carve((size_t)kM * kD * 4);
  int*       nbr   = (int*)carve((size_t)kM * 12 * 4);
  _Float16*  qkvw16 = (_Float16*)carve((size_t)kL * 3 * kD * kD * 2);
  _Float16*  outw16 = (_Float16*)carve((size_t)kL * kD * kD * 2);
  _Float16*  ff1w16 = (_Float16*)carve((size_t)kL * kDFF * kD * 2);
  _Float16*  ff2w16 = (_Float16*)carve((size_t)kL * kD * kDFF * 2);
  _Float16*  gsw16  = (_Float16*)carve((size_t)kG * kD * kD * 2);
  _Float16*  gnw16  = (_Float16*)carve((size_t)kG * kD * kD * 2);
  _Float16*  decw16 = (_Float16*)carve((size_t)kD * kD * 2);

  // 0) one-time weight conversion to f16 (deterministic, every call)
  auto cvtw = [&](const float* src, _Float16* dst, size_t n) {
    f32_to_f16_kernel<<<dim3((unsigned)((n + 255) / 256)), 256, 0, stream>>>(src, dst, (int)n);
  };
  cvtw(t_qkv_w, qkvw16, (size_t)kL * 3 * kD * kD);
  cvtw(t_out_w, outw16, (size_t)kL * kD * kD);
  cvtw(t_ff1_w, ff1w16, (size_t)kL * kDFF * kD);
  cvtw(t_ff2_w, ff2w16, (size_t)kL * kD * kDFF);
  cvtw(g_self_w, gsw16, (size_t)kG * kD * kD);
  cvtw(g_neigh_w, gnw16, (size_t)kG * kD * kD);
  cvtw(dec_w, decw16, (size_t)kD * kD);

  TW tw;
  tw.qkv_w = qkvw16; tw.out_w = outw16; tw.ff1_w = ff1w16; tw.ff2_w = ff2w16;
  tw.qkv_b = t_qkv_b; tw.out_b = t_out_b; tw.ff1_b = t_ff1_b; tw.ff2_b = t_ff2_b;
  tw.ln1_g = t_ln1_g; tw.ln1_b = t_ln1_b; tw.ln2_g = t_ln2_g; tw.ln2_b = t_ln2_b;

  // 1) encoder
  encoder_kernel<<<kM, 128, 0, stream>>>(beats, enc_w1, enc_b1, enc_w2, enc_b2, e);

  // 2) target = temporal(e)
  hipMemcpyAsync(xa, e, (size_t)kM * kD * 4, hipMemcpyDeviceToDevice, stream);
  run_temporal(xa, xb, qkvh, attno, tmp, hff, valid, tw, stream);
  hipMemcpyAsync(out, xa, (size_t)kM * kD * 4, hipMemcpyDeviceToDevice, stream);

  // 3) masked input, x = temporal(e_masked)
  mask_kernel<<<kM, 128, 0, stream>>>(e, node_mask, valid, mask_tok, xa);
  run_temporal(xa, xb, qkvh, attno, tmp, hff, valid, tw, stream);

  // 4) graph: kNN adjacency (top-7 cosine + chain + eye) then 2 conv layers
  float* xg = xa;
  float* xo = xb;
  xnorm_kernel<<<kM, 128, 0, stream>>>(xg, xn);
  topk_kernel<<<dim3(kN, kB), 128, 0, stream>>>(xn, nbr);
  for (int g = 0; g < kG; g++) {
    agg_kernel<<<kM, 128, 0, stream>>>(xg, nbr, aggb);
    gemm_wmma_kernel<float, float, 0, false><<<dim3(kD / 64, kM / 64), 128, 0, stream>>>(
        xg, gsw16 + (size_t)g * kD * kD, g_self_b + g * kD, gout, kM, kD, kD);
    gemm_wmma_kernel<float, float, 0, true><<<dim3(kD / 64, kM / 64), 128, 0, stream>>>(
        aggb, gnw16 + (size_t)g * kD * kD, g_neigh_b + g * kD, gout, kM, kD, kD);
    resln_kernel<<<kM, 128, 0, stream>>>(gout, xg, g_ln_g + g * kD, g_ln_b + g * kD, xo, 1);
    float* t2 = xg; xg = xo; xo = t2;
  }

  // 5) decoder -> recon (second half of d_out)
  gemm_wmma_kernel<float, float, 0, false><<<dim3(kD / 64, kM / 64), 128, 0, stream>>>(
      xg, decw16, dec_b, out + (size_t)kM * kD, kM, kD, kD);

  (void)in_sizes; (void)n_in; (void)out_size; (void)ws_size;
}
